// MultiheadAttentionBase_24610162606373
// MI455X (gfx1250) — matrix operand order repaired and checked by reading, not compiled, not run on hardware
//
#include <hip/hip_runtime.h>
#include <hip/hip_bf16.h>

// Flash attention for MI455X (gfx1250, wave32).
// B=2, H=16, S=2048, D=64, fp32 in/out, causal mask.
//
// Pass 1: convert Q (pre-scaled by 1/sqrt(D)), K to bf16 row-major and V to
//         bf16 *transposed* [bh][d][s] in d_ws (~25.2 MB required).
// Pass 2: flash attention; K/V tiles double-buffered into LDS with
//         global_load_async_to_lds_b128 (ASYNCcnt path, overlapped with
//         compute), scores/PV via v_wmma_f32_16x16x32_bf16, softmax
//         reductions via DPP16.  Masking only in the diagonal epilogue.

typedef __bf16 bf16;
typedef __attribute__((ext_vector_type(16))) __bf16 v16bf;
typedef __attribute__((ext_vector_type(8)))  __bf16 v8bf;
typedef __attribute__((ext_vector_type(8)))  float  v8f;

#define S_LEN 2048
#define D_DIM 64
#define BH    32          // B*H
#define QTILE 64          // query rows per workgroup (4 waves x 16)
#define KTILE 64          // key rows per inner iteration
#define SOFT_SCALE 0.125f // 1/sqrt(64)
#define MASK_VAL (-10000.0f)

// LDS row strides padded to multiples of 16 halves (32B): fragment ds_loads
// stay 32B-aligned contiguous, async b128 stores stay 16B-aligned.
#define KSTRIDE 80   // sK[buf][key][d]  : 64 d + pad    (row = 160B)
#define VSTRIDE 80   // sVT[buf][d][key] : 64 keys + pad (row = 160B)
#define PSTRIDE 64   // sP[wave][row][key]

__device__ __forceinline__ v16bf ld16(const bf16* p) {
    return *reinterpret_cast<const v16bf*>(p);
}

__device__ __forceinline__ v8f wmma_bf16(v16bf a, v16bf b, v8f c) {
    return __builtin_amdgcn_wmma_f32_16x16x32_bf16(
        /*neg_a=*/false, a, /*neg_b=*/false, b,
        /*c_mod=*/(short)0, c, /*reuse_a=*/false, /*reuse_b=*/false);
}

// ---- gfx1250 async global->LDS copy (ASYNCcnt-tracked), SADDR/GVS form ---
__device__ __forceinline__ void async_ld16(unsigned lds_off,
                                           unsigned long long gbase,
                                           unsigned goff) {
    asm volatile("global_load_async_to_lds_b128 %0, %1, %2"
                 :
                 : "v"(lds_off), "v"(goff), "s"(gbase)
                 : "memory");
}
// Wait until ASYNCcnt <= 0 / <= 8 (per-wave, in-order completion).
__device__ __forceinline__ void wait_async0() {
    asm volatile("s_wait_asynccnt 0" ::: "memory");
}
__device__ __forceinline__ void wait_async8() {
    asm volatile("s_wait_asynccnt 8" ::: "memory");
}

// ---- DPP16 cross-lane reductions over each 16-lane half ------------------
// quad_perm(1,0,3,2)=0xB1 (lane^1), quad_perm(2,3,0,1)=0x4E (lane^2),
// row_ror:4=0x124, row_ror:8=0x128.
template <int CTRL>
__device__ __forceinline__ float dppf(float x) {
    return __builtin_bit_cast(float,
        __builtin_amdgcn_update_dpp(0, __builtin_bit_cast(int, x),
                                    CTRL, 0xF, 0xF, true));
}
__device__ __forceinline__ float rowmax16(float x) {
    x = fmaxf(x, dppf<0xB1>(x));
    x = fmaxf(x, dppf<0x4E>(x));
    x = fmaxf(x, dppf<0x124>(x));
    x = fmaxf(x, dppf<0x128>(x));
    return x;
}
__device__ __forceinline__ float rowsum16(float x) {
    x += dppf<0xB1>(x);
    x += dppf<0x4E>(x);
    x += dppf<0x124>(x);
    x += dppf<0x128>(x);
    return x;
}

// ---- One 64-key flash-attention step (MASKED only on diagonal block) -----
template <bool MASKED>
__device__ __forceinline__ void attn_step(
    const bf16 (*__restrict__ sKb)[KSTRIDE],
    const bf16 (*__restrict__ sVTb)[VSTRIDE],
    bf16 (*__restrict__ sPw)[PSTRIDE],
    const v16bf (&qf)[2], v8f (&accv)[4],
    float (&mrow)[8], float (&lrow)[8],
    int kk0, int qfirst, int m, int hi, int hi8) {
    // Scores: four 16x16 tiles, accumulated over d (2 WMMAs each).
    v8f sc[4];
#pragma unroll
    for (int g = 0; g < 4; ++g) {
        v8f z = {};
        z = wmma_bf16(qf[0], ld16(&sKb[g * 16 + m][0  + hi * 16]), z);
        z = wmma_bf16(qf[1], ld16(&sKb[g * 16 + m][32 + hi * 16]), z);
        sc[g] = z;
    }

    // Online softmax (DPP16 reductions; C/D layout row = hi8+r, col = m).
#pragma unroll
    for (int r = 0; r < 8; ++r) {
        const int qr = qfirst + hi8 + r;
        float s[4];
#pragma unroll
        for (int g = 0; g < 4; ++g) {
            s[g] = sc[g][r];
            if (MASKED && (kk0 + g * 16 + m > qr)) s[g] = MASK_VAL;
        }
        float mx = fmaxf(fmaxf(s[0], s[1]), fmaxf(s[2], s[3]));
        mx = rowmax16(mx);
        mx = fmaxf(mx, mrow[r]);
        const float corr = __expf(mrow[r] - mx);
        mrow[r] = mx;
        float p[4], rs = 0.0f;
#pragma unroll
        for (int g = 0; g < 4; ++g) {
            p[g] = __expf(s[g] - mx);
            rs += p[g];
        }
        rs = rowsum16(rs);
        lrow[r] = lrow[r] * corr + rs;
        accv[0][r] *= corr;
        accv[1][r] *= corr;
        accv[2][r] *= corr;
        accv[3][r] *= corr;
#pragma unroll
        for (int g = 0; g < 4; ++g) {
            sPw[hi8 + r][g * 16 + m] = (bf16)p[g];   // C layout -> LDS
        }
    }

    // Re-load P in A layout, accumulate P x V (8 WMMAs).
    const bf16* prow = sPw[m];
    v8bf a00 = *(const v8bf*)(prow + hi * 16);
    v8bf a01 = *(const v8bf*)(prow + hi * 16 + 8);
    v8bf a10 = *(const v8bf*)(prow + 32 + hi * 16);
    v8bf a11 = *(const v8bf*)(prow + 40 + hi * 16);
    v16bf pa0 = __builtin_shufflevector(a00, a01,
        0, 1, 2, 3, 4, 5, 6, 7, 8, 9, 10, 11, 12, 13, 14, 15);
    v16bf pa1 = __builtin_shufflevector(a10, a11,
        0, 1, 2, 3, 4, 5, 6, 7, 8, 9, 10, 11, 12, 13, 14, 15);
#pragma unroll
    for (int t = 0; t < 4; ++t) {
        v16bf vb0 = ld16(&sVTb[t * 16 + m][0  + hi * 16]);
        accv[t] = wmma_bf16(pa0, vb0, accv[t]);
        v16bf vb1 = ld16(&sVTb[t * 16 + m][32 + hi * 16]);
        accv[t] = wmma_bf16(pa1, vb1, accv[t]);
    }
}

// ========================= Pass 1: convert/pack ===========================
__global__ void __launch_bounds__(256)
cvt_pack(const float* __restrict__ Q, const float* __restrict__ K,
         const float* __restrict__ V,
         bf16* __restrict__ Qbf, bf16* __restrict__ Kbf,
         bf16* __restrict__ Vt) {
    const size_t base = ((size_t)blockIdx.x * 256 + threadIdx.x) * 4;

    float4 qv = *(const float4*)(Q + base);
    float4 kv = *(const float4*)(K + base);
    float4 vv = *(const float4*)(V + base);

    Qbf[base + 0] = (bf16)(qv.x * SOFT_SCALE);
    Qbf[base + 1] = (bf16)(qv.y * SOFT_SCALE);
    Qbf[base + 2] = (bf16)(qv.z * SOFT_SCALE);
    Qbf[base + 3] = (bf16)(qv.w * SOFT_SCALE);

    Kbf[base + 0] = (bf16)kv.x;
    Kbf[base + 1] = (bf16)kv.y;
    Kbf[base + 2] = (bf16)kv.z;
    Kbf[base + 3] = (bf16)kv.w;

    // V transpose: [bh][s][d] -> [bh][d][s]
    const size_t bh  = base / ((size_t)S_LEN * D_DIM);
    const int    key = (int)((base / D_DIM) & (S_LEN - 1));
    const int    d0  = (int)(base & (D_DIM - 1));   // 4 consecutive d
    bf16* vt = Vt + bh * (size_t)D_DIM * S_LEN + (size_t)d0 * S_LEN + key;
    vt[0 * S_LEN] = (bf16)vv.x;
    vt[1 * S_LEN] = (bf16)vv.y;
    vt[2 * S_LEN] = (bf16)vv.z;
    vt[3 * S_LEN] = (bf16)vv.w;
}

// ========================= Pass 2: attention ==============================
__global__ void __launch_bounds__(128)
fa_causal_bf16_wmma(const bf16* __restrict__ Qbf,
                    const bf16* __restrict__ Kbf,
                    const bf16* __restrict__ Vt,
                    float* __restrict__ O) {
    __shared__ __align__(32) bf16 sK[2][KTILE][KSTRIDE];   // [buf][key][d]
    __shared__ __align__(32) bf16 sVT[2][D_DIM][VSTRIDE];  // [buf][d][key]
    __shared__ __align__(32) bf16 sP[4][16][PSTRIDE];      // per-wave P scratch

    const int tid  = threadIdx.x;
    const int wave = tid >> 5;
    const int lane = tid & 31;
    const int m    = lane & 15;       // col within 16 / A-row index
    const int hi   = lane >> 4;       // lane half
    const int hi8  = hi << 3;

    const int qblk = blockIdx.x & (S_LEN / QTILE - 1);     // 32 q-blocks
    const int bh   = blockIdx.x >> 5;                      // S/QTILE == 32
    const int q0   = qblk * QTILE;

    const bf16* Qp  = Qbf + (size_t)bh * S_LEN * D_DIM;
    const bf16* Kp  = Kbf + (size_t)bh * S_LEN * D_DIM;
    const bf16* Vtp = Vt  + (size_t)bh * D_DIM * S_LEN;
    float*      Op  = O   + (size_t)bh * S_LEN * D_DIM;

    const int qfirst = q0 + wave * 16;    // first query row of this wave

    // Issue one tile's async copies (8 instructions / wave, 16B each lane).
    auto stage = [&](int buf, int kk0) {
        const unsigned long long kbase =
            (unsigned long long)(uintptr_t)(Kp + (size_t)kk0 * D_DIM);
        const unsigned long long vbase =
            (unsigned long long)(uintptr_t)(Vtp + kk0);
#pragma unroll
        for (int i = 0; i < 4; ++i) {
            const int idx = tid + i * 128;      // 512 x 16B per tensor
            const int row = idx >> 3;           // 0..63
            const int seg = idx & 7;            // 16B segment within row
            async_ld16((unsigned)(uintptr_t)&sK[buf][row][seg * 8],
                       kbase, (unsigned)((row * D_DIM + seg * 8) * 2));
            async_ld16((unsigned)(uintptr_t)&sVT[buf][row][seg * 8],
                       vbase, (unsigned)((row * S_LEN + seg * 8) * 2));
        }
    };

    // ---- Load Q (pre-scaled bf16) into two 16x32 A-fragments ------------
    const bf16* qrp = Qp + (size_t)(qfirst + m) * D_DIM;
    v16bf qf[2];
#pragma unroll
    for (int part = 0; part < 2; ++part) {
        v8bf lo = *(const v8bf*)(qrp + part * 32 + hi * 16);
        v8bf hp = *(const v8bf*)(qrp + part * 32 + hi * 16 + 8);
        qf[part] = __builtin_shufflevector(lo, hp,
            0, 1, 2, 3, 4, 5, 6, 7, 8, 9, 10, 11, 12, 13, 14, 15);
    }

    // ---- Online-softmax state --------------------------------------------
    float mrow[8], lrow[8];
    v8f accv[4];
#pragma unroll
    for (int r = 0; r < 8; ++r) { mrow[r] = -1e30f; lrow[r] = 0.0f; }
#pragma unroll
    for (int t = 0; t < 4; ++t) { v8f z = {}; accv[t] = z; }

    const int num_kb = (q0 + QTILE) / KTILE;   // causal bound == qblk + 1

    stage(0, 0);   // preload first tile

    for (int kb = 0; kb < num_kb; ++kb) {
        const int kk0 = kb * KTILE;
        const int cur = kb & 1;

        if (kb + 1 < num_kb) {
            // Issue next tile into the other buffer (its readers finished at
            // the end-of-iteration barrier of kb-1), then wait for the
            // current tile's 8 copies (in-order per-wave completion).
            stage(cur ^ 1, kk0 + KTILE);
            wait_async8();
            if (kb + 2 < num_kb) {   // gfx1250 global_prefetch_b8, 2 ahead
                __builtin_prefetch(Kp + (size_t)(kk0 + 2 * KTILE) * D_DIM
                                   + tid * 32, 0, 1);
                __builtin_prefetch(Vtp + (size_t)(tid >> 1) * S_LEN
                                   + kk0 + 2 * KTILE + (tid & 1) * 32, 0, 1);
            }
        } else {
            wait_async0();
        }
        __syncthreads();   // all waves' current-tile copies visible

        if (kb == num_kb - 1) {
            attn_step<true >(sK[cur], sVT[cur], sP[wave], qf, accv,
                             mrow, lrow, kk0, qfirst, m, hi, hi8);
        } else {
            attn_step<false>(sK[cur], sVT[cur], sP[wave], qf, accv,
                             mrow, lrow, kk0, qfirst, m, hi, hi8);
        }
        __syncthreads();   // readers done before buffer is rewritten
    }

    // ---- Normalize and write fp32 output --------------------------------
#pragma unroll
    for (int r = 0; r < 8; ++r) {
        const int qr = qfirst + hi8 + r;
        const float inv = 1.0f / lrow[r];
        float* orow = Op + (size_t)qr * D_DIM;
#pragma unroll
        for (int t = 0; t < 4; ++t) {
            orow[t * 16 + m] = accv[t][r] * inv;
        }
    }
}

extern "C" void kernel_launch(void* const* d_in, const int* in_sizes, int n_in,
                              void* d_out, int out_size, void* d_ws, size_t ws_size,
                              hipStream_t stream) {
    (void)in_sizes; (void)n_in; (void)out_size; (void)ws_size;
    const float* q = (const float*)d_in[0];
    const float* k = (const float*)d_in[1];
    const float* v = (const float*)d_in[2];
    // d_in[3] is the causal mask; it is static (tril) and baked into the kernel.
    float* out = (float*)d_out;

    const size_t N = (size_t)BH * S_LEN * D_DIM;   // 4.19M elements per tensor
    bf16* Qbf = (bf16*)d_ws;                       // needs 3*N*2 = ~25.2 MB
    bf16* Kbf = Qbf + N;
    bf16* Vt  = Kbf + N;

    cvt_pack<<<dim3((unsigned)(N / 4 / 256)), dim3(256), 0, stream>>>(
        q, k, v, Qbf, Kbf, Vt);

    dim3 grid(BH * (S_LEN / QTILE));   // 32 * 32 = 1024 workgroups
    dim3 block(128);                   // 4 wave32 waves
    fa_causal_bf16_wmma<<<grid, block, 0, stream>>>(Qbf, Kbf, Vt, out);
}
